// EventDrivenSSMLayer_66529043415048
// MI455X (gfx1250) — matrix operand order repaired and checked by reading, not compiled, not run on hardware
//
#include <hip/hip_runtime.h>

// ---------------------------------------------------------------------------
// Event-driven spiking SSM layer for MI455X (gfx1250, wave32, WMMA).
//
// x[8,32,256,512] f32, A[64,64], B[64,512], C[512,64], D[512,512].
// R = B*S = 2048 rows per timestep; sequential scan over T=32 via
// stream-ordered launches (graph-capture safe).
//
// Roofline: 43.6 GFLOP total (x@D^T dominates), 256 MB HBM floor ~11us at
// 23.3 TB/s. f16 WMMA (v_wmma_f32_16x16x32_f16, f32 accum) keeps this
// memory-bound; f32 WMMA (16x16x4) would be ~8x compute-bound. LIF state
// (sv/ov/thresholds) stays f32.
//
// v2: out-kernel is M-blocked 2x — each wave computes two 16x16 tiles that
// share every D/C B-fragment in registers, halving L2->WGP weight traffic
// (64->32 MB/step) and doubling WMMA density per wave.
// ---------------------------------------------------------------------------

typedef __attribute__((ext_vector_type(16))) _Float16 v16h;
typedef __attribute__((ext_vector_type(8)))  _Float16 v8h;
typedef __attribute__((ext_vector_type(8)))  float    v8f;
typedef __attribute__((ext_vector_type(4)))  float    v4f;
typedef __attribute__((ext_vector_type(4)))  _Float16 v4h;

constexpr int   Bb = 8, T = 32, S = 256, DM = 512, DS = 64;
constexpr int   R  = Bb * S;                       // 2048 rows per timestep
constexpr float MEM_DECAY = 0.60653065971263342f;  // exp(-1/2)
constexpr float ADAPT     = 0.1f;
constexpr float INV_R     = 1.0f / (float)R;

// ---- WMMA fragment loaders -------------------------------------------------
// A-matrix 16x32 f16 layout (ISA 7.12.2): lanes 0-15 hold K 0..7 & 16..23,
// lanes 16-31 hold K 8..15 & 24..31, row M = lane%16.  Two 16B loads.
__device__ inline v16h load_a_frag(const _Float16* rowbase, int kbase, int hi) {
  const _Float16* p = rowbase + kbase + hi * 8;
  v8h lo = *(const v8h*)(p);
  v8h hv = *(const v8h*)(p + 16);
  return __builtin_shufflevector(lo, hv, 0,1,2,3,4,5,6,7,8,9,10,11,12,13,14,15);
}
// B-matrix 32x16 f16 layout: lane (N=lane%16) holds K 0..7 & 16..23 (hi=0)
// or K 8..15 & 24..31 (hi=1), matching the A fragment.  Two 16B loads.
__device__ inline v16h load_b_frag(const _Float16* rowbase, int kbase, int hi) {
  const _Float16* p = rowbase + kbase + hi * 8;
  v8h lo = *(const v8h*)(p);
  v8h hv = *(const v8h*)(p + 16);
  return __builtin_shufflevector(lo, hv, 0,1,2,3,4,5,6,7,8,9,10,11,12,13,14,15);
}
__device__ inline v8f wmma16(v16h a, v16h b, v8f c) {
  return __builtin_amdgcn_wmma_f32_16x16x32_f16(false, a, false, b,
                                                (short)0, c, false, false);
}

// ---- one-time init / conversion --------------------------------------------
__global__ void init_kernel(float* sv, float* ov, _Float16* h16,
                            float* s_thr, float* o_thr,
                            float* s_cnt, float* o_cnt, int* flags) {
  size_t tid = (size_t)blockIdx.x * blockDim.x + threadIdx.x;
  if (tid < (size_t)R * DM) ov[tid] = 0.f;
  if (tid < (size_t)R * DS) { sv[tid] = 0.f; h16[tid] = (_Float16)0.f; }
  if (tid < DS) { s_thr[tid] = 1.0f; s_cnt[tid] = 0.f; }
  if (tid < DM) { o_thr[tid] = 1.0f; o_cnt[tid] = 0.f; }
  if (tid < T)  flags[tid] = 0;
}

// 4 elements per thread: b128 load, b64 store (pure streaming pass, 192 MB).
__global__ void cvt_x_kernel(const float* __restrict__ x,
                             _Float16* __restrict__ x16,
                             int* __restrict__ flags) {
  size_t idx = ((size_t)blockIdx.x * blockDim.x + threadIdx.x) * 4;
  v4f v = *(const v4f*)(x + idx);
  v4h h;
  h.x = (_Float16)v.x; h.y = (_Float16)v.y;
  h.z = (_Float16)v.z; h.w = (_Float16)v.w;
  *(v4h*)(x16 + idx) = h;
  if (v.x > 0.f || v.y > 0.f || v.z > 0.f || v.w > 0.f) {
    int t = (int)((idx >> 17) & 31);   // idx / (S*DM) % T  (4-group same t)
    flags[t] = 1;                      // benign same-value race
  }
}

__global__ void cvt_w_kernel(const float* __restrict__ A, const float* __restrict__ Bw,
                             const float* __restrict__ C, const float* __restrict__ D,
                             _Float16* A16, _Float16* B16, _Float16* C16, _Float16* D16) {
  int idx = blockIdx.x * blockDim.x + threadIdx.x;
  if (idx < DS * DS) A16[idx] = (_Float16)A[idx];
  if (idx < DS * DM) { B16[idx] = (_Float16)Bw[idx]; C16[idx] = (_Float16)C[idx]; }
  if (idx < DM * DM) D16[idx] = (_Float16)D[idx];
}

// ---- per-timestep: state update + state LIF ---------------------------------
// grid 128 (row tiles of 16), block 128 (4 waves; wave w -> d_state cols 16w..)
__global__ void state_kernel(const _Float16* __restrict__ x16,
                             const _Float16* __restrict__ A16,
                             const _Float16* __restrict__ B16,
                             _Float16* __restrict__ h16,
                             float* __restrict__ sv,
                             const float* __restrict__ s_thr,
                             float* __restrict__ s_cnt,
                             const int* __restrict__ flags, int t) {
  const int lane = threadIdx.x & 31;
  const int wave = threadIdx.x >> 5;
  const int hi   = lane >> 4;
  const int m    = lane & 15;
  const int row0 = blockIdx.x * 16;
  const int n0   = wave * 16;
  const int r    = row0 + m;

  // state_transition = h @ A^T   (K = 64)
  const _Float16* hrow = h16 + (size_t)r * DS;
  const _Float16* arow = A16 + (size_t)(n0 + m) * DS;
  v8f acc = {};
  #pragma unroll
  for (int k = 0; k < DS; k += 32)
    acc = wmma16(load_a_frag(hrow, k, hi), load_b_frag(arow, k, hi), acc);

  // all waves done reading h before anyone writes h_new (same 16 rows shared)
  __syncthreads();

  if (flags[t]) {                       // uniform branch: EXEC stays all-ones
    const size_t xoff = (((size_t)(r >> 8) * T + t) * S + (size_t)(r & 255)) * DM;
    const _Float16* xrow = x16 + xoff;
    const _Float16* brow = B16 + (size_t)(n0 + m) * DM;
    #pragma unroll
    for (int k = 0; k < DM; k += 32) {
      __builtin_prefetch(xrow + k + 128, 0, 3);   // WGP-scope prefetch
      acc = wmma16(load_a_frag(xrow, k, hi), load_b_frag(brow, k, hi), acc);
    }
  }

  // LIF: v_pot = sv*decay + u ; spike ; reset ; h_new = spike (f16 operand)
  const int n = n0 + m;
  const float thr = s_thr[n];
  float cnt = 0.f;
  #pragma unroll
  for (int v = 0; v < 8; ++v) {
    const int re = row0 + v + 8 * hi;             // C/D layout: M = v + 8*hi
    const size_t idx = (size_t)re * DS + n;
    float vpot = sv[idx] * MEM_DECAY + acc[v];
    float sp   = (vpot - thr >= 0.f) ? 1.f : 0.f;
    sv[idx]    = vpot * (1.f - sp);
    h16[idx]   = (_Float16)sp;
    cnt += sp;
  }
  atomicAdd(&s_cnt[n], cnt);
}

// ---- per-timestep: output update + output LIF -------------------------------
// grid (64,4), block 256 (8 waves). Each wave: 2 row-tiles x 1 n-tile,
// sharing every C/D B-fragment between the two accumulators (2x reuse).
__global__ void out_kernel(const _Float16* __restrict__ x16,
                           const _Float16* __restrict__ C16,
                           const _Float16* __restrict__ D16,
                           const _Float16* __restrict__ h16,
                           float* __restrict__ ov,
                           const float* __restrict__ o_thr,
                           float* __restrict__ o_cnt,
                           const int* __restrict__ flags,
                           float* __restrict__ out, int t) {
  const int lane = threadIdx.x & 31;
  const int wave = threadIdx.x >> 5;
  const int hi   = lane >> 4;
  const int m    = lane & 15;
  const int row0 = blockIdx.x * 32;                 // 2 row-tiles per wave
  const int n0   = blockIdx.y * 128 + wave * 16;
  const int n    = n0 + m;
  const int r0   = row0 + m;
  const int r1   = row0 + 16 + m;

  if (!flags[t]) {                       // inactive: out_spike = 0, ov/o_thr kept
    #pragma unroll
    for (int j = 0; j < 2; ++j)
      #pragma unroll
      for (int v = 0; v < 8; ++v) {
        const int re = row0 + 16 * j + v + 8 * hi;
        const size_t o = (((size_t)(re >> 8) * T + t) * S + (size_t)(re & 255)) * DM + n;
        out[o] = 0.f;
      }
    return;
  }

  // out_update = h_new @ C^T (K=64) + x_t @ D^T (K=512)
  const _Float16* h0   = h16 + (size_t)r0 * DS;
  const _Float16* h1   = h16 + (size_t)r1 * DS;
  const _Float16* crow = C16 + (size_t)n * DS;
  v8f acc0 = {}, acc1 = {};
  #pragma unroll
  for (int k = 0; k < DS; k += 32) {
    v16h bf = load_b_frag(crow, k, hi);
    acc0 = wmma16(load_a_frag(h0, k, hi), bf, acc0);
    acc1 = wmma16(load_a_frag(h1, k, hi), bf, acc1);
  }

  const size_t xo0 = (((size_t)(r0 >> 8) * T + t) * S + (size_t)(r0 & 255)) * DM;
  const size_t xo1 = (((size_t)(r1 >> 8) * T + t) * S + (size_t)(r1 & 255)) * DM;
  const _Float16* x0   = x16 + xo0;
  const _Float16* x1   = x16 + xo1;
  const _Float16* drow = D16 + (size_t)n * DM;
  #pragma unroll
  for (int k = 0; k < DM; k += 32) {
    __builtin_prefetch(x0 + k + 128, 0, 3);
    __builtin_prefetch(drow + k + 128, 0, 3);
    v16h bf = load_b_frag(drow, k, hi);     // shared by both row-tiles
    acc0 = wmma16(load_a_frag(x0, k, hi), bf, acc0);
    acc1 = wmma16(load_a_frag(x1, k, hi), bf, acc1);
  }

  const float thr = o_thr[n];
  float cnt = 0.f;
  #pragma unroll
  for (int j = 0; j < 2; ++j) {
    const v8f& acc = j ? acc1 : acc0;
    #pragma unroll
    for (int v = 0; v < 8; ++v) {
      const int re = row0 + 16 * j + v + 8 * hi;
      const size_t idx = (size_t)re * DM + n;
      float vpot = ov[idx] * MEM_DECAY + acc[v];
      float sp   = (vpot - thr >= 0.f) ? 1.f : 0.f;
      ov[idx]    = vpot * (1.f - sp);
      const size_t o = (((size_t)(re >> 8) * T + t) * S + (size_t)(re & 255)) * DM + n;
      out[o] = sp;
      cnt += sp;
    }
  }
  atomicAdd(&o_cnt[n], cnt);
}

// ---- per-timestep: adaptive threshold update --------------------------------
__global__ void thr_kernel(float* s_thr, float* s_cnt, float* o_thr, float* o_cnt,
                           const int* __restrict__ flags, int t) {
  const int i = threadIdx.x;
  if (i < DS) {
    s_thr[i] += ADAPT * (s_cnt[i] * INV_R - 0.1f);   // always updated
    s_cnt[i]  = 0.f;
  }
  if (i < DM) {
    if (flags[t]) o_thr[i] += ADAPT * (o_cnt[i] * INV_R - 0.1f);
    o_cnt[i] = 0.f;
  }
}

// ---------------------------------------------------------------------------
extern "C" void kernel_launch(void* const* d_in, const int* in_sizes, int n_in,
                              void* d_out, int out_size, void* d_ws, size_t ws_size,
                              hipStream_t stream) {
  const float* x  = (const float*)d_in[0];
  const float* A  = (const float*)d_in[1];
  const float* Bw = (const float*)d_in[2];
  const float* C  = (const float*)d_in[3];
  const float* D  = (const float*)d_in[4];
  float* out = (float*)d_out;
  char*  ws  = (char*)d_ws;

  // workspace layout (~69.4 MB)
  _Float16* x16   = (_Float16*)(ws);                 // 67,108,864 B
  _Float16* A16   = (_Float16*)(ws + 67108864);      //      8,192 B
  _Float16* B16   = (_Float16*)(ws + 67117056);      //     65,536 B
  _Float16* C16   = (_Float16*)(ws + 67182592);      //     65,536 B
  _Float16* D16   = (_Float16*)(ws + 67248128);      //    524,288 B
  _Float16* h16   = (_Float16*)(ws + 67772416);      //    262,144 B
  float*    sv    = (float*)   (ws + 68034560);      //    524,288 B
  float*    ov    = (float*)   (ws + 68558848);      //  4,194,304 B
  float*    s_thr = (float*)   (ws + 72753152);      //        256 B
  float*    o_thr = (float*)   (ws + 72753408);      //      2,048 B
  float*    s_cnt = (float*)   (ws + 72755456);      //        256 B
  float*    o_cnt = (float*)   (ws + 72755712);      //      2,048 B
  int*      flags = (int*)     (ws + 72757760);      //        128 B

  init_kernel<<<dim3(4096), dim3(256), 0, stream>>>(sv, ov, h16, s_thr, o_thr,
                                                    s_cnt, o_cnt, flags);
  cvt_x_kernel<<<dim3(32768), dim3(256), 0, stream>>>(x, x16, flags);
  cvt_w_kernel<<<dim3(1024), dim3(256), 0, stream>>>(A, Bw, C, D,
                                                     A16, B16, C16, D16);
  for (int t = 0; t < T; ++t) {
    state_kernel<<<dim3(128),   dim3(128), 0, stream>>>(x16, A16, B16, h16, sv,
                                                        s_thr, s_cnt, flags, t);
    out_kernel  <<<dim3(64, 4), dim3(256), 0, stream>>>(x16, C16, D16, h16, ov,
                                                        o_thr, o_cnt, flags, out, t);
    thr_kernel  <<<dim3(1),     dim3(512), 0, stream>>>(s_thr, s_cnt,
                                                        o_thr, o_cnt, flags, t);
  }
}
